// Net_1151051235463
// MI455X (gfx1250) — matrix operand order repaired and checked by reading, not compiled, not run on hardware
//
#include <hip/hip_runtime.h>
#include <hip/hip_bf16.h>
#include <math.h>

#define N_NODES_C  10000
#define NUM_GENE_C 2000
#define N_EDGES_C  160000
#define HEADS_C    10
#define HID_C      128
#define LATENT_C   128
#define LOWER_C    64
#define CLUSTERS_C 15
#define TOTAL_E_C  (N_EDGES_C + N_NODES_C)
#define LDS_STRIDE 40                 // 32 + 8 pad: 16B-aligned rows, conflict-free gathers
#define TILE_SH    (128 * LDS_STRIDE) // ushorts per stage

typedef __attribute__((ext_vector_type(16))) __bf16       v16bf;
typedef __attribute__((ext_vector_type(8)))  float        v8f;
typedef __attribute__((ext_vector_type(8)))  unsigned int v8u;

// ---------------- helpers ----------------
__device__ __forceinline__ unsigned short f32_to_bf16(float f) {
  unsigned int u = __float_as_uint(f);
  u += 0x7FFFu + ((u >> 16) & 1u);          // round-to-nearest-even
  return (unsigned short)(u >> 16);
}
__device__ __forceinline__ unsigned int pack2bf(float a, float b) {
  return (unsigned int)f32_to_bf16(a) | ((unsigned int)f32_to_bf16(b) << 16);
}
__device__ __forceinline__ float wave_reduce_sum(float v) {
  for (int off = 16; off > 0; off >>= 1) v += __shfl_xor(v, off, 32);
  return v;
}
__device__ __forceinline__ void atomicMaxFloat(float* addr, float val) {
  if (val >= 0.0f) atomicMax((int*)addr, __float_as_int(val));
  else             atomicMin((unsigned int*)addr, __float_as_uint(val));
}

// ---------------- WMMA GEMM (double-buffered LDS, 1 barrier per k-tile) ----------------
// A tile: 128x32, fp32 global -> packed bf16 regs (fused augmentation) -> LDS [m][k]
__device__ __forceinline__ void stage_load_A(const float* __restrict__ A,
    const float* __restrict__ noise, const float* __restrict__ invnrm,
    int M, int K, int aug, int m0, int k0, int tid, uint4 out[2])
{
  #pragma unroll
  for (int it = 0; it < 2; ++it) {
    int id = tid + it * 256;
    int r  = id >> 2;              // 0..127
    int cb = (id & 3) << 3;        // 0,8,16,24
    int gr = m0 + r;
    float v[8];
    if (gr < M && (k0 + cb + 7) < K) {
      const float* ap = &A[(size_t)gr * K + k0 + cb];
      float4 a0 = *reinterpret_cast<const float4*>(ap);
      float4 a1 = *reinterpret_cast<const float4*>(ap + 4);
      v[0] = a0.x; v[1] = a0.y; v[2] = a0.z; v[3] = a0.w;
      v[4] = a1.x; v[5] = a1.y; v[6] = a1.z; v[7] = a1.w;
      if (aug) {
        const float* np = &noise[(size_t)gr * K + k0 + cb];
        float4 n0v = *reinterpret_cast<const float4*>(np);
        float4 n1v = *reinterpret_cast<const float4*>(np + 4);
        float iw = invnrm[gr];
        v[0] += n0v.x * iw; v[1] += n0v.y * iw; v[2] += n0v.z * iw; v[3] += n0v.w * iw;
        v[4] += n1v.x * iw; v[5] += n1v.y * iw; v[6] += n1v.z * iw; v[7] += n1v.w * iw;
      }
    } else {
      #pragma unroll
      for (int e = 0; e < 8; ++e) {
        int gc = k0 + cb + e;
        float t = 0.0f;
        if (gr < M && gc < K) {
          t = A[(size_t)gr * K + gc];
          if (aug) t += noise[(size_t)gr * K + gc] * invnrm[gr];
        }
        v[e] = t;
      }
    }
    uint4 q;
    q.x = pack2bf(v[0], v[1]); q.y = pack2bf(v[2], v[3]);
    q.z = pack2bf(v[4], v[5]); q.w = pack2bf(v[6], v[7]);
    out[it] = q;
  }
}
__device__ __forceinline__ void stage_store_A(unsigned short* __restrict__ As,
                                              int tid, const uint4 in[2])
{
  #pragma unroll
  for (int it = 0; it < 2; ++it) {
    int id = tid + it * 256;
    int r  = id >> 2;
    int cb = (id & 3) << 3;
    *reinterpret_cast<uint4*>(&As[r * LDS_STRIDE + cb]) = in[it];
  }
}

// B tile: 32x128 fp32 global -> packed bf16 regs -> LDS transposed [n][k]
__device__ __forceinline__ void stage_load_B(const float* __restrict__ B,
    int N, int K, int n0, int k0, int tid, uint4 out[2])
{
  #pragma unroll
  for (int it = 0; it < 2; ++it) {
    int id = tid + it * 256;
    int kr = id >> 4;              // 0..31
    int nb = (id & 15) << 3;       // 0..120
    int gk = k0 + kr;
    float v[8];
    if (gk < K) {                  // N is always a multiple of 128 here
      const float* bp = &B[(size_t)gk * N + n0 + nb];
      float4 b0v = *reinterpret_cast<const float4*>(bp);
      float4 b1v = *reinterpret_cast<const float4*>(bp + 4);
      v[0] = b0v.x; v[1] = b0v.y; v[2] = b0v.z; v[3] = b0v.w;
      v[4] = b1v.x; v[5] = b1v.y; v[6] = b1v.z; v[7] = b1v.w;
    } else {
      #pragma unroll
      for (int e = 0; e < 8; ++e) v[e] = 0.0f;
    }
    uint4 q;
    q.x = pack2bf(v[0], v[1]); q.y = pack2bf(v[2], v[3]);
    q.z = pack2bf(v[4], v[5]); q.w = pack2bf(v[6], v[7]);
    out[it] = q;
  }
}
__device__ __forceinline__ void stage_store_B(unsigned short* __restrict__ Bs,
                                              int tid, const uint4 in[2])
{
  #pragma unroll
  for (int it = 0; it < 2; ++it) {
    int id = tid + it * 256;
    int kr = id >> 4;
    int nb = (id & 15) << 3;
    unsigned int qq[4] = {in[it].x, in[it].y, in[it].z, in[it].w};
    #pragma unroll
    for (int e = 0; e < 8; ++e)
      Bs[(nb + e) * LDS_STRIDE + kr] = (unsigned short)(qq[e >> 1] >> ((e & 1) * 16));
  }
}

// C[M,N] = A'[M,K] @ B[K,N]   (A,B fp32 in HBM, bf16 in LDS, f32 accumulate)
// aug:      A' = A + noise * invnrm[row]   (fused x_aug)
// epilogue: 0 = none, 1 = bias+relu, 2 = bias
__global__ void __launch_bounds__(256)
gemm_bf16_wmma(const float* __restrict__ A, const float* __restrict__ noise,
               const float* __restrict__ invnrm,
               const float* __restrict__ B, float* __restrict__ C,
               const float* __restrict__ bias,
               int M, int N, int K, int aug, int epilogue)
{
  __shared__ __align__(16) unsigned short As[2 * TILE_SH]; // ping-pong [m][k]
  __shared__ __align__(16) unsigned short Bs[2 * TILE_SH]; // ping-pong transposed [n][k]

  const int tid  = threadIdx.x;
  const int wave = tid >> 5;
  const int lane = tid & 31;
  const int lg   = lane >> 4;        // lane group 0/1
  const int lr   = lane & 15;
  const int m0   = blockIdx.y * 128;
  const int n0   = blockIdx.x * 128;
  const int wm   = wave & 1;         // 2 wave-rows   (64 rows each)
  const int wn   = wave >> 1;        // 4 wave-cols   (32 cols each)

  v8f acc[4][2];
  {
    v8f z = {};
    for (int i = 0; i < 4; ++i) for (int j = 0; j < 2; ++j) acc[i][j] = z;
  }

  const int ktiles = (K + 31) >> 5;
  uint4 ra[2], rb[2];
  stage_load_A(A, noise, invnrm, M, K, aug, m0, 0, tid, ra);
  stage_load_B(B, N, K, n0, 0, tid, rb);

  for (int kt = 0; kt < ktiles; ++kt) {
    unsigned short* as = As + (kt & 1) * TILE_SH;
    unsigned short* bs = Bs + (kt & 1) * TILE_SH;
    // stores target the buffer whose last readers were fenced by the PREVIOUS
    // iteration's barrier; the other buffer is being read concurrently -> safe.
    stage_store_A(as, tid, ra);
    stage_store_B(bs, tid, rb);
    __syncthreads();                       // single barrier per k-tile

    if (kt + 1 < ktiles) {                 // global prefetch overlaps the WMMA burst
      int k0 = (kt + 1) << 5;
      stage_load_A(A, noise, invnrm, M, K, aug, m0, k0, tid, ra);
      stage_load_B(B, N, K, n0, k0, tid, rb);
    }

    const unsigned int* As32 = reinterpret_cast<const unsigned int*>(as);
    const unsigned int* Bs32 = reinterpret_cast<const unsigned int*>(bs);

    // preload ALL fragments, then issue the 8 WMMAs back-to-back
    // B frag (32x16 bf16): lanes 0-15 K=0..15, lanes 16-31 K=16..31; N = lane&15
    v16bf bfrag[2];
    #pragma unroll
    for (int nj = 0; nj < 2; ++nj) {
      int n = wn * 32 + nj * 16 + lr;
      v8u u;
      #pragma unroll
      for (int vv = 0; vv < 8; ++vv) {
        int k = lg * 16 + 2 * vv;
        u[vv] = Bs32[(n * LDS_STRIDE + k) >> 1];
      }
      bfrag[nj] = __builtin_bit_cast(v16bf, u);
    }
    // A frag (16x32 bf16): lanes 0-15 K=0..15, lanes 16-31 K=16..31; M = lane&15
    v16bf afrag[4];
    #pragma unroll
    for (int mi = 0; mi < 4; ++mi) {
      int m = wm * 64 + mi * 16 + lr;
      v8u u;
      #pragma unroll
      for (int vv = 0; vv < 8; ++vv) {
        int k = lg * 16 + 2 * vv;
        u[vv] = As32[(m * LDS_STRIDE + k) >> 1];
      }
      afrag[mi] = __builtin_bit_cast(v16bf, u);
    }
    #pragma unroll
    for (int mi = 0; mi < 4; ++mi) {
      #pragma unroll
      for (int nj = 0; nj < 2; ++nj) {
        acc[mi][nj] = __builtin_amdgcn_wmma_f32_16x16x32_bf16(
            false, afrag[mi], false, bfrag[nj], (short)0, acc[mi][nj], false, false);
      }
    }
    // no tail barrier: next iteration writes the OTHER LDS stage, and writes to
    // THIS stage only occur after the next iteration's barrier.
  }

  // ---- epilogue: C/D layout: VGPR r -> M = r + 8*lanegroup, N = lane&15 ----
  for (int mi = 0; mi < 4; ++mi) {
    for (int nj = 0; nj < 2; ++nj) {
      int n = n0 + wn * 32 + nj * 16 + lr;
      #pragma unroll
      for (int rr = 0; rr < 8; ++rr) {
        int m = m0 + wm * 64 + mi * 16 + lg * 8 + rr;
        if (m < M && n < N) {
          float v = acc[mi][nj][rr];
          if (epilogue) {
            v += bias[n];
            if (epilogue == 1) v = fmaxf(v, 0.0f);
          }
          C[(size_t)m * N + n] = v;
        }
      }
    }
  }
}

// ---------------- attention (segment softmax + scatter) ----------------
__global__ void __launch_bounds__(256)
row_invnorm(const float* __restrict__ noise, float* __restrict__ invnrm, int M, int K) {
  int wid  = (int)((blockIdx.x * (size_t)blockDim.x + threadIdx.x) >> 5);
  int lane = threadIdx.x & 31;
  if (wid >= M) return;
  float s = 0.0f;
  const float* r = noise + (size_t)wid * K;
  for (int c = lane; c < K; c += 32) { float v = r[c]; s += v * v; }
  s = wave_reduce_sum(s);
  if (lane == 0) invnrm[wid] = 1.0f / fmaxf(sqrtf(s), 1e-12f);
}

// per-(node,head) wave: es/ed dot products; init segment max/sum
__global__ void __launch_bounds__(256)
att_scores(const float* __restrict__ h, const float* __restrict__ a_src,
           const float* __restrict__ a_dst, float* __restrict__ es,
           float* __restrict__ ed, float* __restrict__ mmax, float* __restrict__ ssum,
           int n_nodes, int H, int D)
{
  int wid  = (int)((blockIdx.x * (size_t)blockDim.x + threadIdx.x) >> 5);
  int lane = threadIdx.x & 31;
  if (wid >= n_nodes * H) return;
  int n = wid / H, hh = wid % H;
  const float* hr = h + (size_t)n * H * D + (size_t)hh * D;
  float s1 = 0.0f, s2 = 0.0f;
  for (int d = lane; d < D; d += 32) {
    float v = hr[d];
    s1 += v * a_src[hh * D + d];
    s2 += v * a_dst[hh * D + d];
  }
  s1 = wave_reduce_sum(s1);
  s2 = wave_reduce_sum(s2);
  if (lane == 0) { es[wid] = s1; ed[wid] = s2; mmax[wid] = -3.0e38f; ssum[wid] = 0.0f; }
}

__device__ __forceinline__ void edge_sd(const int* ei, int e, int& s, int& d) {
  if (e < N_EDGES_C) { s = ei[e]; d = ei[N_EDGES_C + e]; }
  else               { s = d = e - N_EDGES_C; }            // self loops
}

__global__ void __launch_bounds__(256)
att_edge_alpha(const int* __restrict__ ei, const float* __restrict__ es,
               const float* __restrict__ ed, float* __restrict__ w,
               float* __restrict__ mmax, int H)
{
  size_t idx = blockIdx.x * (size_t)blockDim.x + threadIdx.x;
  if (idx >= (size_t)TOTAL_E_C * H) return;
  int e = (int)(idx / H), hh = (int)(idx % H);
  int s, d; edge_sd(ei, e, s, d);
  float a = es[s * H + hh] + ed[d * H + hh];
  a = (a > 0.0f) ? a : 0.2f * a;                 // leaky_relu(0.2)
  w[idx] = a;
  atomicMaxFloat(&mmax[d * H + hh], a);
}

__global__ void __launch_bounds__(256)
att_edge_exp(const int* __restrict__ ei, float* __restrict__ w,
             const float* __restrict__ mmax, float* __restrict__ ssum, int H)
{
  size_t idx = blockIdx.x * (size_t)blockDim.x + threadIdx.x;
  if (idx >= (size_t)TOTAL_E_C * H) return;
  int e = (int)(idx / H), hh = (int)(idx % H);
  int s, d; edge_sd(ei, e, s, d); (void)s;
  float p = __expf(w[idx] - mmax[d * H + hh]);
  w[idx] = p;
  atomicAdd(&ssum[d * H + hh], p);
}

// per-(edge,head) wave: out[dst,h,:] += h[src,h,:] * (p / s[dst,h])
__global__ void __launch_bounds__(256)
att_scatter(const int* __restrict__ ei, const float* __restrict__ w,
            const float* __restrict__ ssum, const float* __restrict__ h,
            float* __restrict__ out, int H, int D)
{
  size_t wid = (blockIdx.x * (size_t)blockDim.x + threadIdx.x) >> 5;
  int lane = threadIdx.x & 31;
  if (wid >= (size_t)TOTAL_E_C * H) return;
  int e = (int)(wid / H), hh = (int)(wid % H);
  int s, d; edge_sd(ei, e, s, d);
  float wgt = w[wid] / ssum[d * H + hh];
  const float* hs = h + (size_t)s * H * D + (size_t)hh * D;
  float* od = out + (size_t)d * H * D + (size_t)hh * D;
  for (int dd = lane; dd < D; dd += 32) atomicAdd(&od[dd], hs[dd] * wgt);
}

// ---------------- elementwise / reductions ----------------
__global__ void __launch_bounds__(256)
fill_zero(float* __restrict__ p, long long n) {
  for (long long i = blockIdx.x * (long long)blockDim.x + threadIdx.x; i < n;
       i += (long long)gridDim.x * blockDim.x) p[i] = 0.0f;
}

__global__ void __launch_bounds__(256)
bias_relu_(float* __restrict__ x, const float* __restrict__ b, int M, int N) {
  size_t idx = blockIdx.x * (size_t)blockDim.x + threadIdx.x;
  if (idx >= (size_t)M * N) return;
  int n = (int)(idx % N);
  x[idx] = fmaxf(x[idx] + b[n], 0.0f);
}

__global__ void __launch_bounds__(256)
mean_heads_relu(const float* __restrict__ agg, const float* __restrict__ b,
                float* __restrict__ gx, int M, int H, int D)
{
  size_t idx = blockIdx.x * (size_t)blockDim.x + threadIdx.x;
  if (idx >= (size_t)M * D) return;
  int n = (int)(idx / D), d = (int)(idx % D);
  float s = 0.0f;
  for (int hh = 0; hh < H; ++hh) s += agg[(size_t)n * H * D + (size_t)hh * D + d];
  gx[idx] = fmaxf(s * (1.0f / H) + b[d], 0.0f);
}

__global__ void col_sum(const float* __restrict__ x, float* __restrict__ g, int M, int D) {
  int d = threadIdx.x;                 // blockDim.x == D (128)
  int r0 = blockIdx.x * 256;
  int r1 = r0 + 256; if (r1 > M) r1 = M;
  float s = 0.0f;
  for (int r = r0; r < r1; ++r) s += x[(size_t)r * D + d];
  atomicAdd(&g[d], s);
}

__global__ void loss_kernel(const float* g0, const float* g1, const float* g2, float* out) {
  float d01 = 0, d02 = 0, d12 = 0;
  const float inv = 1.0f / N_NODES_C;
  for (int i = 0; i < LATENT_C; ++i) {
    float a = g0[i] * inv, b = g1[i] * inv, c = g2[i] * inv;
    d01 += a * b; d02 += a * c; d12 += b * c;
  }
  float e01 = __expf(d01 * 5.0f), e02 = __expf(d02 * 5.0f), e12 = __expf(d12 * 5.0f);
  out[0] = -log10f(e01 / (e01 + e02 + e12));
}

// fused small projections: node head (LATENT->LOWER) and cluster head (LATENT->CLUSTERS, transposed)
__global__ void __launch_bounds__(256)
proj_kernel(const float* __restrict__ gx, const float* __restrict__ Wf,
            const float* __restrict__ bf_, const float* __restrict__ Wc,
            const float* __restrict__ bc_, float* __restrict__ out_node,
            float* __restrict__ out_c, int do_c)
{
  int JT = LOWER_C + (do_c ? CLUSTERS_C : 0);
  size_t idx = blockIdx.x * (size_t)blockDim.x + threadIdx.x;
  if (idx >= (size_t)N_NODES_C * JT) return;
  int n = (int)(idx / JT), j = (int)(idx % JT);
  const float* g = gx + (size_t)n * LATENT_C;
  if (j < LOWER_C) {
    float s = bf_[j];
    for (int k = 0; k < LATENT_C; ++k) s += g[k] * Wf[k * LOWER_C + j];
    out_node[(size_t)n * LOWER_C + j] = s;
  } else {
    int jc = j - LOWER_C;
    float s = bc_[jc];
    for (int k = 0; k < LATENT_C; ++k) s += g[k] * Wc[k * CLUSTERS_C + jc];
    out_c[(size_t)jc * N_NODES_C + n] = s;
  }
}

// ---------------- host orchestration ----------------
static inline unsigned int cdiv_u(long long a, long long b) {
  return (unsigned int)((a + b - 1) / b);
}

extern "C" void kernel_launch(void* const* d_in, const int* in_sizes, int n_in,
                              void* d_out, int out_size, void* d_ws, size_t ws_size,
                              hipStream_t stream)
{
  const float* x   = (const float*)d_in[0];
  const int*   ei  = (const int*)  d_in[1];
  const float* noi = (const float*)d_in[2];
  const float* W1  = (const float*)d_in[3];
  const float* a1s = (const float*)d_in[4];
  const float* a1d = (const float*)d_in[5];
  const float* b1  = (const float*)d_in[6];
  const float* W2  = (const float*)d_in[7];
  const float* a2s = (const float*)d_in[8];
  const float* a2d = (const float*)d_in[9];
  const float* b2  = (const float*)d_in[10];
  const float* Wm1 = (const float*)d_in[11];
  const float* bm1 = (const float*)d_in[12];
  const float* Wm2 = (const float*)d_in[13];
  const float* bm2 = (const float*)d_in[14];
  const float* Wf  = (const float*)d_in[15];
  const float* bf_ = (const float*)d_in[16];
  const float* Wc  = (const float*)d_in[17];
  const float* bc_ = (const float*)d_in[18];
  float* out = (float*)d_out;

  // workspace layout (floats)
  float* ws = (float*)d_ws;
  size_t o = 0;
  float* hbuf = ws + o; o += (size_t)N_NODES_C * HEADS_C * HID_C;   // 12.8M
  float* obuf = ws + o; o += (size_t)N_NODES_C * HEADS_C * HID_C;   // 12.8M
  float* gxb  = ws + o; o += (size_t)N_NODES_C * LATENT_C;          // 1.28M
  float* tb   = ws + o; o += (size_t)N_NODES_C * LATENT_C;          // 1.28M
  float* es   = ws + o; o += (size_t)N_NODES_C * HEADS_C;
  float* ed   = ws + o; o += (size_t)N_NODES_C * HEADS_C;
  float* mm   = ws + o; o += (size_t)N_NODES_C * HEADS_C;
  float* ss   = ws + o; o += (size_t)N_NODES_C * HEADS_C;
  float* we   = ws + o; o += (size_t)TOTAL_E_C * HEADS_C;           // 1.7M
  float* inr  = ws + o; o += (size_t)N_NODES_C;
  float* gsum = ws + o; o += 3 * LATENT_C;

  // output offsets (return order, flat)
  float* out_n0 = out;
  float* out_n1 = out + (size_t)N_NODES_C * LOWER_C;
  float* out_n2 = out + 2 * (size_t)N_NODES_C * LOWER_C;
  float* out_ls = out + 3 * (size_t)N_NODES_C * LOWER_C;
  float* out_c0 = out_ls + 1;
  float* out_c1 = out_c0 + (size_t)CLUSTERS_C * N_NODES_C;

  dim3 blk(256);
  const long long FD = (long long)N_NODES_C * HEADS_C * HID_C;  // 12.8M
  const long long NW = (long long)N_NODES_C * HEADS_C;          // node-head pairs
  const long long EH = (long long)TOTAL_E_C * HEADS_C;          // edge-head pairs

  fill_zero<<<1, blk, 0, stream>>>(gsum, 3 * LATENT_C);
  row_invnorm<<<cdiv_u((long long)N_NODES_C * 32, 256), blk, 0, stream>>>(noi, inr, N_NODES_C, NUM_GENE_C);

  auto gat_branch = [&](int aug, float* gxout) {
    // layer 1: h = x' @ W1  [N, H*HID]
    dim3 g1(cdiv_u(HEADS_C * HID_C, 128), cdiv_u(N_NODES_C, 128));
    gemm_bf16_wmma<<<g1, blk, 0, stream>>>(x, noi, inr, W1, hbuf, nullptr,
                                           N_NODES_C, HEADS_C * HID_C, NUM_GENE_C, aug, 0);
    att_scores<<<cdiv_u(NW * 32, 256), blk, 0, stream>>>(hbuf, a1s, a1d, es, ed, mm, ss,
                                                         N_NODES_C, HEADS_C, HID_C);
    fill_zero<<<4096, blk, 0, stream>>>(obuf, FD);
    att_edge_alpha<<<cdiv_u(EH, 256), blk, 0, stream>>>(ei, es, ed, we, mm, HEADS_C);
    att_edge_exp  <<<cdiv_u(EH, 256), blk, 0, stream>>>(ei, we, mm, ss, HEADS_C);
    att_scatter<<<cdiv_u(EH * 32, 256), blk, 0, stream>>>(ei, we, ss, hbuf, obuf, HEADS_C, HID_C);
    bias_relu_<<<cdiv_u(FD, 256), blk, 0, stream>>>(obuf, b1, N_NODES_C, HEADS_C * HID_C);

    // layer 2: h2 = h1 @ W2  [N, H*LATENT]  (obuf -> hbuf, then aggregate back into obuf)
    dim3 g2(cdiv_u(HEADS_C * LATENT_C, 128), cdiv_u(N_NODES_C, 128));
    gemm_bf16_wmma<<<g2, blk, 0, stream>>>(obuf, nullptr, nullptr, W2, hbuf, nullptr,
                                           N_NODES_C, HEADS_C * LATENT_C, HEADS_C * HID_C, 0, 0);
    att_scores<<<cdiv_u(NW * 32, 256), blk, 0, stream>>>(hbuf, a2s, a2d, es, ed, mm, ss,
                                                         N_NODES_C, HEADS_C, LATENT_C);
    fill_zero<<<4096, blk, 0, stream>>>(obuf, FD);
    att_edge_alpha<<<cdiv_u(EH, 256), blk, 0, stream>>>(ei, es, ed, we, mm, HEADS_C);
    att_edge_exp  <<<cdiv_u(EH, 256), blk, 0, stream>>>(ei, we, mm, ss, HEADS_C);
    att_scatter<<<cdiv_u(EH * 32, 256), blk, 0, stream>>>(ei, we, ss, hbuf, obuf, HEADS_C, LATENT_C);
    mean_heads_relu<<<cdiv_u((long long)N_NODES_C * LATENT_C, 256), blk, 0, stream>>>(
        obuf, b2, gxout, N_NODES_C, HEADS_C, LATENT_C);
  };

  // branch 0: gx
  gat_branch(0, gxb);
  col_sum<<<cdiv_u(N_NODES_C, 256), dim3(LATENT_C), 0, stream>>>(gxb, gsum, N_NODES_C, LATENT_C);
  proj_kernel<<<cdiv_u((long long)N_NODES_C * (LOWER_C + CLUSTERS_C), 256), blk, 0, stream>>>(
      gxb, Wf, bf_, Wc, bc_, out_n0, out_c0, 1);

  // branch 1: gx1 (augmented input, fused in GEMM staging)
  gat_branch(1, gxb);
  col_sum<<<cdiv_u(N_NODES_C, 256), dim3(LATENT_C), 0, stream>>>(gxb, gsum + LATENT_C, N_NODES_C, LATENT_C);
  proj_kernel<<<cdiv_u((long long)N_NODES_C * (LOWER_C + CLUSTERS_C), 256), blk, 0, stream>>>(
      gxb, Wf, bf_, Wc, bc_, out_n1, out_c1, 1);

  // branch 2: MLP  gx2 = relu(x@Wm1+bm1)@Wm2 + bm2
  {
    dim3 gm1(cdiv_u(LATENT_C, 128), cdiv_u(N_NODES_C, 128));
    gemm_bf16_wmma<<<gm1, blk, 0, stream>>>(x, nullptr, nullptr, Wm1, tb, bm1,
                                            N_NODES_C, LATENT_C, NUM_GENE_C, 0, 1);
    gemm_bf16_wmma<<<gm1, blk, 0, stream>>>(tb, nullptr, nullptr, Wm2, gxb, bm2,
                                            N_NODES_C, LATENT_C, LATENT_C, 0, 2);
    col_sum<<<cdiv_u(N_NODES_C, 256), dim3(LATENT_C), 0, stream>>>(gxb, gsum + 2 * LATENT_C,
                                                                   N_NODES_C, LATENT_C);
    proj_kernel<<<cdiv_u((long long)N_NODES_C * LOWER_C, 256), blk, 0, stream>>>(
        gxb, Wf, bf_, Wc, bc_, out_n2, nullptr, 0);
  }

  // contrastive graph loss scalar
  loss_kernel<<<1, 1, 0, stream>>>(gsum, gsum + LATENT_C, gsum + 2 * LATENT_C, out_ls);
}
